// CosineDistMult_MS_79250736546779
// MI455X (gfx1250) — matrix operand order repaired and checked by reading, not compiled, not run on hardware
//
#include <hip/hip_runtime.h>
#include <hip/hip_bf16.h>
#include <math.h>

typedef __attribute__((ext_vector_type(16))) _Float16 v16h;
typedef __attribute__((ext_vector_type(8)))  _Float16 v8h;
typedef __attribute__((ext_vector_type(8)))  float    v8f;

#define BB   512
#define DD   768
#define RR   128
#define NCH  160            // max padded 16-col chunks: ceil bound (512+128*15)/16=152 -> 160
#define PMAX (NCH*16)       // 2560 padded gathered columns

#define MS_ALPHA 2.0f
#define MS_BETA  50.0f
#define MS_BASE  0.5f
#define MS_EPS   0.1f

// ---------------- workspace layout (bytes) ----------------
static constexpr size_t OFF_CNT  = 0;                       // RR ints
static constexpr size_t OFF_PST  = 512;                     // RR ints
static constexpr size_t OFF_CREL = 1024;                    // NCH ints
static constexpr size_t OFF_BMAP = 2048;                    // PMAX ints
static constexpr size_t OFF_XP   = 16384;                   // f16 [DD][PMAX]
static constexpr size_t SZ_XP    = (size_t)DD * PMAX * 2;   // 3,932,160
static constexpr size_t OFF_T    = OFF_XP + SZ_XP;          // f32 [BB][DD]
static constexpr size_t SZ_T     = (size_t)BB * DD * 4;
static constexpr size_t OFF_TNH  = OFF_T + SZ_T;            // f16 [BB][DD]
static constexpr size_t SZ_TNH   = (size_t)BB * DD * 2;
static constexpr size_t OFF_C1T  = OFF_TNH + SZ_TNH;        // f16 [DD][BB]
static constexpr size_t SZ_C1T   = (size_t)DD * BB * 2;
static constexpr size_t OFF_MAT  = OFF_C1T + SZ_C1T;        // f32 [BB][BB]
static constexpr size_t SZ_MAT   = (size_t)BB * BB * 4;
static constexpr size_t OFF_RL   = OFF_MAT + SZ_MAT;        // f32 [BB]
// total ~8.2 MB

// ---------------- kernel 1: group rows by relation ----------------
// Builds: cnt[r], pstart[r] (multiple of 16), crel[chunk] (-1 = dead chunk),
// bmap[p] = original batch row for padded column p (-1 = padding).
__global__ void k_group(const int* __restrict__ re_id,
                        int* cnt, int* pstart, int* crel, int* bmap) {
  int tid = threadIdx.x;                       // blockDim = 128
  for (int i = tid; i < NCH;  i += 128) crel[i] = -1;
  for (int i = tid; i < PMAX; i += 128) bmap[i] = -1;
  __syncthreads();
  if (tid < RR) {
    int c = 0;
    for (int b = 0; b < BB; ++b) if (re_id[b] == tid) ++c;
    cnt[tid] = c;
  }
  __syncthreads();
  if (tid == 0) {
    int off = 0, ch = 0;
    for (int r = 0; r < RR; ++r) {
      pstart[r] = off;
      int nc = (cnt[r] + 15) >> 4;
      for (int c = 0; c < nc; ++c) crel[ch++] = r;
      off += nc << 4;                          // keeps pstart 16-aligned => chunk base = chunk*16
    }
  }
  __syncthreads();
  if (tid < RR) {
    int idx = pstart[tid];
    for (int b = 0; b < BB; ++b) if (re_id[b] == tid) bmap[idx++] = b;
  }
}

// ---------------- kernel 2: gathered, transposed f16 copy of cui_0 ----------------
// Xp[k][p] = cui_0[bmap[p]][k]  (0 for padding). K-major so WMMA B-frag loads are contiguous.
__global__ void k_xp(const float* __restrict__ cui0, const int* __restrict__ bmap,
                     _Float16* __restrict__ Xp) {
  int idx = blockIdx.x * 256 + threadIdx.x;    // grid = DD*PMAX/256 exactly
  int k = idx / PMAX, p = idx - k * PMAX;
  int b = bmap[p];
  Xp[idx] = (b >= 0) ? (_Float16)cui0[(size_t)b * DD + k] : (_Float16)0.0f;
}

// ---------------- kernel 3: normalize cui_1 rows -> f16, K-major ----------------
__global__ void k_normC1(const float* __restrict__ cui1, _Float16* __restrict__ C1t) {
  __shared__ float red[256];
  int b = blockIdx.x, tid = threadIdx.x;
  float ss = 0.f;
  for (int i = tid; i < DD; i += 256) { float v = cui1[(size_t)b * DD + i]; ss += v * v; }
  red[tid] = ss; __syncthreads();
  for (int s = 128; s > 0; s >>= 1) { if (tid < s) red[tid] += red[tid + s]; __syncthreads(); }
  float inv = 1.0f / fmaxf(sqrtf(red[0]), 1e-12f);
  for (int i = tid; i < DD; i += 256)
    C1t[(size_t)i * BB + b] = (_Float16)(cui1[(size_t)b * DD + i] * inv);
}

// ---------------- kernel 4: grouped GEMM  T[b][i] = sum_j M_r[i][j] * x_b[j] ----------------
// One wave per (mtile, chunk): 16x16 f32 tile, K-loop 24 x v_wmma_f32_16x16x32_f16.
// This is the 302 MB HBM stream; each used relation matrix is read ~once.
__global__ void k_phase2(const float* __restrict__ transform,
                         const _Float16* __restrict__ Xp,
                         const int* __restrict__ crel, const int* __restrict__ bmap,
                         float* __restrict__ T) {
  int wave = threadIdx.x >> 5;
  int lane = threadIdx.x & 31;
  int mtile = blockIdx.x * 8 + wave;           // 0..47 (blockIdx.x 0..5)
  int chunk = blockIdx.y;                      // 0..NCH-1
  int r = crel[chunk];
  if (r < 0) return;                           // wave-uniform: EXEC all-ones through WMMA
  int cbase = chunk * 16;
  int kshift = (lane & 16) ? 8 : 0;            // A K-half selector; doubles as D M-half offset
  const float* A0 = transform + (size_t)r * DD * DD + (size_t)(mtile * 16 + (lane & 15)) * DD;
  v8f acc = {};
  for (int k0 = 0; k0 < DD; k0 += 32) {
    int koff = k0 + kshift;
    float4 f0 = *(const float4*)(A0 + koff);
    float4 f1 = *(const float4*)(A0 + koff + 4);
    float4 f2 = *(const float4*)(A0 + koff + 16);
    float4 f3 = *(const float4*)(A0 + koff + 20);
    v16h a;
    a[0]=(_Float16)f0.x;  a[1]=(_Float16)f0.y;  a[2]=(_Float16)f0.z;  a[3]=(_Float16)f0.w;
    a[4]=(_Float16)f1.x;  a[5]=(_Float16)f1.y;  a[6]=(_Float16)f1.z;  a[7]=(_Float16)f1.w;
    a[8]=(_Float16)f2.x;  a[9]=(_Float16)f2.y;  a[10]=(_Float16)f2.z; a[11]=(_Float16)f2.w;
    a[12]=(_Float16)f3.x; a[13]=(_Float16)f3.y; a[14]=(_Float16)f3.z; a[15]=(_Float16)f3.w;
    v16h bf;  // lane=N, elements=K (table order)
    for (int s = 0; s < 16; ++s) bf[s] = Xp[(size_t)(koff + (s & 7) + ((s >> 3) << 4)) * PMAX + cbase + (lane & 15)];
    acc = __builtin_amdgcn_wmma_f32_16x16x32_f16(false, a, false, bf,
                                                 (short)0, acc, false, false);
  }
  int b = bmap[cbase + (lane & 15)];
  if (b >= 0) {                                // padding columns dropped
    float* dst = T + (size_t)b * DD + mtile * 16 + kshift;   // 8 contiguous M values per lane
    *(v8f*)dst = acc;
  }
}

// ---------------- kernel 5: normalize T rows -> f16 row-major ----------------
__global__ void k_normT(const float* __restrict__ T, _Float16* __restrict__ Tnh) {
  __shared__ float red[256];
  int b = blockIdx.x, tid = threadIdx.x;
  float ss = 0.f;
  for (int i = tid; i < DD; i += 256) { float v = T[(size_t)b * DD + i]; ss += v * v; }
  red[tid] = ss; __syncthreads();
  for (int s = 128; s > 0; s >>= 1) { if (tid < s) red[tid] += red[tid + s]; __syncthreads(); }
  float inv = 1.0f / fmaxf(sqrtf(red[0]), 1e-12f);
  for (int i = tid; i < DD; i += 256)
    Tnh[(size_t)b * DD + i] = (_Float16)(T[(size_t)b * DD + i] * inv);
}

// ---------------- kernel 6: similarity GEMM  mat = Tn @ C1n^T (512x512, K=768) ----------------
__global__ void k_phase3(const _Float16* __restrict__ Tnh, const _Float16* __restrict__ C1t,
                         float* __restrict__ mat) {
  int wave = threadIdx.x >> 5;
  int lane = threadIdx.x & 31;
  int mtile = blockIdx.x * 8 + wave;           // 0..31 (blockIdx.x 0..3)
  int ntile = blockIdx.y;                      // 0..31
  int kshift = (lane & 16) ? 8 : 0;
  const _Float16* Ar = Tnh + (size_t)(mtile * 16 + (lane & 15)) * DD;
  v8f acc = {};
  for (int k0 = 0; k0 < DD; k0 += 32) {
    int koff = k0 + kshift;
    v8h h0 = *(const v8h*)(Ar + koff);         // K koff..koff+7
    v8h h1 = *(const v8h*)(Ar + koff + 16);    // K koff+16..koff+23
    v16h a;
    a[0]=h0[0]; a[1]=h0[1]; a[2]=h0[2]; a[3]=h0[3];
    a[4]=h0[4]; a[5]=h0[5]; a[6]=h0[6]; a[7]=h0[7];
    a[8]=h1[0]; a[9]=h1[1]; a[10]=h1[2]; a[11]=h1[3];
    a[12]=h1[4]; a[13]=h1[5]; a[14]=h1[6]; a[15]=h1[7];
    v16h bf;
    for (int s = 0; s < 16; ++s) bf[s] = C1t[(size_t)(koff + (s & 7) + ((s >> 3) << 4)) * BB + ntile * 16 + (lane & 15)];
    acc = __builtin_amdgcn_wmma_f32_16x16x32_f16(false, a, false, bf,
                                                 (short)0, acc, false, false);
  }
  int col = ntile * 16 + (lane & 15);
  int rbase = mtile * 16 + kshift;
  #pragma unroll
  for (int v = 0; v < 8; ++v)
    mat[(size_t)(rbase + v) * BB + col] = acc[v];
}

// ---------------- kernel 7: MultiSimilarity miner + loss per row ----------------
__device__ __forceinline__ void lse_push(float& m, float& s, float x) {
  if (x <= m) { s += __expf(x - m); }
  else        { s = s * __expf(m - x) + 1.0f; m = x; }   // m==-inf: exp(-inf)=0 -> s=1
}
__device__ __forceinline__ void lse_merge(float& m, float& s, float m2, float s2) {
  if (m2 == -INFINITY || s2 == 0.0f) return;
  if (m == -INFINITY) { m = m2; s = s2; return; }
  if (m2 <= m) { s += s2 * __expf(m2 - m); }
  else         { s = s * __expf(m - m2) + s2; m = m2; }
}

__global__ void k_rowloss(const float* __restrict__ mat, const int* __restrict__ lab1,
                          float* __restrict__ rowloss) {
  __shared__ float srow[BB];
  __shared__ int   slab[BB];
  __shared__ float redA[256];
  __shared__ float redB[256];
  int a = blockIdx.x, tid = threadIdx.x;
  for (int j = tid; j < BB; j += 256) { srow[j] = mat[(size_t)a * BB + j]; slab[j] = lab1[j]; }
  __syncthreads();
  int la = slab[a];
  // pass 1: max over negatives, min over positives (diagonal included, as in reference)
  float mx = -INFINITY, mn = INFINITY;
  for (int j = tid; j < BB; j += 256) {
    float v = srow[j];
    if (slab[j] == la) mn = fminf(mn, v); else mx = fmaxf(mx, v);
  }
  redA[tid] = mx; redB[tid] = mn; __syncthreads();
  for (int s = 128; s > 0; s >>= 1) {
    if (tid < s) { redA[tid] = fmaxf(redA[tid], redA[tid + s]);
                   redB[tid] = fminf(redB[tid], redB[tid + s]); }
    __syncthreads();
  }
  float max_neg = redA[0], min_pos = redB[0];
  __syncthreads();
  // pass 2: masked stable logsumexp (online per-thread, tree merge)
  float pm = -INFINITY, ps = 0.f, nm = -INFINITY, ns = 0.f;
  for (int j = tid; j < BB; j += 256) {
    float v = srow[j];
    if (slab[j] == la) { if (v - MS_EPS < max_neg) lse_push(pm, ps, MS_ALPHA * (MS_BASE - v)); }
    else               { if (v + MS_EPS > min_pos) lse_push(nm, ns, MS_BETA  * (v - MS_BASE)); }
  }
  redA[tid] = pm; redB[tid] = ps; __syncthreads();
  for (int s = 128; s > 0; s >>= 1) {
    if (tid < s) lse_merge(redA[tid], redB[tid], redA[tid + s], redB[tid + s]);
    __syncthreads();
  }
  float posL = 0.f;
  if (tid == 0) {
    float m = redA[0], s = redB[0];
    lse_push(m, s, 0.0f);                      // the "add_one"; empty mask => log(1)=0
    posL = m + logf(s);
  }
  __syncthreads();
  redA[tid] = nm; redB[tid] = ns; __syncthreads();
  for (int s = 128; s > 0; s >>= 1) {
    if (tid < s) lse_merge(redA[tid], redB[tid], redA[tid + s], redB[tid + s]);
    __syncthreads();
  }
  if (tid == 0) {
    float m = redA[0], s = redB[0];
    lse_push(m, s, 0.0f);
    float negL = m + logf(s);
    rowloss[a] = posL * (1.0f / MS_ALPHA) + negL * (1.0f / MS_BETA);
  }
}

// ---------------- kernel 8: mean ----------------
__global__ void k_final(const float* __restrict__ rowloss, float* __restrict__ out) {
  __shared__ float red[256];
  int tid = threadIdx.x;
  float s = 0.f;
  for (int i = tid; i < BB; i += 256) s += rowloss[i];
  red[tid] = s; __syncthreads();
  for (int st = 128; st > 0; st >>= 1) { if (tid < st) red[tid] += red[tid + st]; __syncthreads(); }
  if (tid == 0) out[0] = red[0] * (1.0f / (float)BB);
}

extern "C" void kernel_launch(void* const* d_in, const int* in_sizes, int n_in,
                              void* d_out, int out_size, void* d_ws, size_t ws_size,
                              hipStream_t stream) {
  const float* cui0      = (const float*)d_in[0];
  const float* cui1      = (const float*)d_in[1];
  const int*   re_id     = (const int*)  d_in[2];
  // d_in[3] = cui_label_0 : unused by reference
  const int*   lab1      = (const int*)  d_in[4];
  const float* transform = (const float*)d_in[5];
  float* out = (float*)d_out;

  char* w = (char*)d_ws;
  int*      cnt    = (int*)(w + OFF_CNT);
  int*      pstart = (int*)(w + OFF_PST);
  int*      crel   = (int*)(w + OFF_CREL);
  int*      bmap   = (int*)(w + OFF_BMAP);
  _Float16* Xp     = (_Float16*)(w + OFF_XP);
  float*    T      = (float*)(w + OFF_T);
  _Float16* Tnh    = (_Float16*)(w + OFF_TNH);
  _Float16* C1t    = (_Float16*)(w + OFF_C1T);
  float*    mat    = (float*)(w + OFF_MAT);
  float*    rl     = (float*)(w + OFF_RL);

  k_group  <<<1, 128, 0, stream>>>(re_id, cnt, pstart, crel, bmap);
  k_xp     <<<(DD * PMAX) / 256, 256, 0, stream>>>(cui0, bmap, Xp);
  k_normC1 <<<BB, 256, 0, stream>>>(cui1, C1t);
  k_phase2 <<<dim3(6, NCH), 256, 0, stream>>>(transform, Xp, crel, bmap, T);
  k_normT  <<<BB, 256, 0, stream>>>(T, Tnh);
  k_phase3 <<<dim3(4, 32), 256, 0, stream>>>(Tnh, C1t, mat);
  k_rowloss<<<BB, 256, 0, stream>>>(mat, lab1, rl);
  k_final  <<<1, 256, 0, stream>>>(rl, out);
}
